// S2Layer_68710886801414
// MI455X (gfx1250) — matrix operand order repaired and checked by reading, not compiled, not run on hardware
//
#include <hip/hip_runtime.h>
#include <math.h>

typedef _Float16 half_t;
typedef __attribute__((ext_vector_type(16))) _Float16 v16h;
typedef __attribute__((ext_vector_type(8)))  _Float16 v8h;
typedef __attribute__((ext_vector_type(8)))  float    v8f;

#define LAM 0.1f

// ---------------------------------------------------------------------------
// WMMA helpers (CDNA5 v_wmma_f32_16x16x32_f16, wave32)
// ---------------------------------------------------------------------------
static __device__ __forceinline__ v8f wmma_f16(v16h a, v16h b, v8f c) {
    return __builtin_amdgcn_wmma_f32_16x16x32_f16(
        /*neg_a=*/false, a, /*neg_b=*/false, b,
        /*c_mod=*/(short)0, c, /*reuse_a=*/false, /*reuse_b=*/false);
}

// A fragment 16x32 f16 (ISA 7.12.2): lane L row M=L&15.
// lanes<16 hold K = {0..7, 16..23}+; lanes>=16 hold K = {8..15, 24..31}.
// Both runs are contiguous -> two 16B loads.
static __device__ __forceinline__ v16h load_a_frag(const half_t* rowPtr, int kBase, int lane) {
    int kOff = (lane & 16) ? 8 : 0;
    union { v16h v; v8h h[2]; } u;
    u.h[0] = *(const v8h*)(rowPtr + kBase + kOff);
    u.h[1] = *(const v8h*)(rowPtr + kBase + kOff + 16);
    return u.v;
}

// B fragment from pre-packed layout: 16 contiguous halfs per lane.
static __device__ __forceinline__ v16h load_b_frag(const half_t* pack, int nkc,
                                                   int ct, int kc, int lane) {
    return *(const v16h*)(pack + ((((ct * nkc + kc) * 32) + lane) << 4));
}

// ---------------------------------------------------------------------------
// Pre-pack a f32 [K x N] row-major matrix into WMMA B-fragment order (f16).
// B 32x16 layout: lanes<16 rows K=kc*32+{0..7,16..23}, lanes>=16 rows K=kc*32+{8..15,24..31},
// element i within lane = K low->high, column N = ct*16 + (lane&15).
// ---------------------------------------------------------------------------
__global__ void pack_b_kernel(const float* __restrict__ src, int srcStride, int srcRowOff,
                              int K, int N, half_t* __restrict__ dst) {
    int nkc = K >> 5;
    int total = nkc * (N >> 4) * 512;
    for (int idx = blockIdx.x * blockDim.x + threadIdx.x; idx < total;
         idx += gridDim.x * blockDim.x) {
        int i    = idx & 15;
        int lane = (idx >> 4) & 31;
        int kc   = (idx >> 9) % nkc;
        int ct   = idx / (512 * nkc);
        int k = kc * 32 + ((lane & 16) ? 8 : 0) + (i & 7) + ((i & 8) ? 16 : 0);
        int n = ct * 16 + (lane & 15);
        dst[idx] = (half_t)src[(size_t)(srcRowOff + k) * srcStride + n];
    }
}

__global__ void cvt_h_kernel(const float* __restrict__ src, half_t* __restrict__ dst, long n) {
    long stride = (long)gridDim.x * blockDim.x;
    for (long i = (long)blockIdx.x * blockDim.x + threadIdx.x; i < n; i += stride)
        dst[i] = (half_t)src[i];
}

// semb_contrib[s][n] = a_b1[n] + sum_j sign_emb[s][j] * a_w1[128+j][n]
__global__ void semb_kernel(const float* __restrict__ sign_emb,
                            const float* __restrict__ a_w1,
                            const float* __restrict__ a_b1,
                            float* __restrict__ sembC) {
    int idx = blockIdx.x * blockDim.x + threadIdx.x;
    if (idx >= 3 * 128) return;
    int s = idx >> 7, n = idx & 127;
    float acc = a_b1[n];
    #pragma unroll
    for (int j = 0; j < 8; ++j)
        acc += sign_emb[s * 8 + j] * a_w1[(128 + j) * 128 + n];
    sembC[idx] = acc;
}

// ---------------------------------------------------------------------------
// Node GEMM: V = H@Wv (f32 + f16 copy), T = H@Wt (f16 copy).
// Block = 8 waves; wave w handles output col tile w (0..3 -> V, 4..7 -> T),
// block b handles node rows 16b..16b+15. K = 128 = 4 chunks.
// ---------------------------------------------------------------------------
__global__ __launch_bounds__(256)
void node_gemm_kernel(const half_t* __restrict__ Hh,
                      const half_t* __restrict__ WvPack,
                      const half_t* __restrict__ WtPack,
                      float* __restrict__ V,
                      half_t* __restrict__ Vh,
                      half_t* __restrict__ Th,
                      int n_nodes) {
    int wave = threadIdx.x >> 5;
    int lane = threadIdx.x & 31;
    int n0 = blockIdx.x * 16;
    if (n0 >= n_nodes) return;

    const half_t* rowPtr = Hh + (size_t)(n0 + (lane & 15)) * 128;
    const half_t* pack = (wave < 4) ? WvPack : WtPack;
    int ct = wave & 3;

    v8f acc = {};
    #pragma unroll
    for (int kc = 0; kc < 4; ++kc) {
        v16h a = load_a_frag(rowPtr, kc * 32, lane);
        v16h b = load_b_frag(pack, 4, ct, kc, lane);
        acc = wmma_f16(a, b, acc);
    }
    int col  = ct * 16 + (lane & 15);
    int rOff = (lane & 16) ? 8 : 0;
    #pragma unroll
    for (int r = 0; r < 8; ++r) {
        size_t node = (size_t)(n0 + r + rOff);
        float x = acc[r];
        if (wave < 4) {
            V [node * 64 + col] = x;
            Vh[node * 64 + col] = (half_t)x;
        } else {
            Th[node * 64 + col] = (half_t)x;
        }
    }
}

// ---------------------------------------------------------------------------
// Edge MLP + softshrink + signed scatter. One wave per 16-edge tile.
//   h = relu(t_i@A1t + v_j@A1v + sembC[sign]);  alpha = h@a_w2 + b2
//   c = sign>0 ? ss(alpha) : sign<0 ? -softplus(gamma)*|ss(alpha)| : 0
//   agg[col] += c * V[row];  loss += |ss(alpha)|
// ---------------------------------------------------------------------------
__global__ __launch_bounds__(256)
void edge_mlp_kernel(const int* __restrict__ edge_index,
                     const int* __restrict__ edge_sign,
                     const half_t* __restrict__ Th,
                     const half_t* __restrict__ Vh,
                     const float* __restrict__ V,
                     const half_t* __restrict__ A1tPack,
                     const half_t* __restrict__ A1vPack,
                     const float* __restrict__ sembC,
                     const float* __restrict__ a_w2,
                     const float* __restrict__ a_b2,
                     const float* __restrict__ gamma,
                     float* __restrict__ agg,
                     float* __restrict__ lossAcc,
                     int n_edges) {
    int wave = threadIdx.x >> 5;
    int lane = threadIdx.x & 31;
    long tile = (long)blockIdx.x * 8 + wave;
    long e0 = tile * 16;
    if (e0 >= n_edges) return;

    int m = lane & 15;
    long e = e0 + m;
    if (e >= n_edges) e = n_edges - 1;          // safety clamp (E%16==0 normally)
    int rowIdx = edge_index[e];
    int colIdx = edge_index[(long)n_edges + e];
    int sgn    = edge_sign[e];
    int sIdx   = sgn + 1; sIdx = sIdx < 0 ? 0 : (sIdx > 2 ? 2 : sIdx);

    const half_t* tRow = Th + (size_t)colIdx * 64;
    const half_t* vRow = Vh + (size_t)rowIdx * 64;
    v16h a0 = load_a_frag(tRow, 0,  lane);
    v16h a1 = load_a_frag(tRow, 32, lane);
    v16h a2 = load_a_frag(vRow, 0,  lane);
    v16h a3 = load_a_frag(vRow, 32, lane);

    int rOff = (lane & 16) ? 8 : 0;
    int sM[8];
    #pragma unroll
    for (int r = 0; r < 8; ++r)
        sM[r] = __shfl(sIdx, r + rOff, 32);

    float part[8];
    #pragma unroll
    for (int r = 0; r < 8; ++r) part[r] = 0.0f;

    #pragma unroll
    for (int ct = 0; ct < 8; ++ct) {
        v8f acc;
        #pragma unroll
        for (int r = 0; r < 8; ++r)
            acc[r] = sembC[sM[r] * 128 + ct * 16 + m];
        acc = wmma_f16(a0, load_b_frag(A1tPack, 2, ct, 0, lane), acc);
        acc = wmma_f16(a1, load_b_frag(A1tPack, 2, ct, 1, lane), acc);
        acc = wmma_f16(a2, load_b_frag(A1vPack, 2, ct, 0, lane), acc);
        acc = wmma_f16(a3, load_b_frag(A1vPack, 2, ct, 1, lane), acc);
        float w2 = a_w2[ct * 16 + m];
        #pragma unroll
        for (int r = 0; r < 8; ++r) {
            float h = acc[r] > 0.0f ? acc[r] : 0.0f;
            part[r] += h * w2;
        }
    }
    // row sums live across 16 lanes of each half-wave
    #pragma unroll
    for (int r = 0; r < 8; ++r) {
        float p = part[r];
        p += __shfl_xor(p, 1, 32);
        p += __shfl_xor(p, 2, 32);
        p += __shfl_xor(p, 4, 32);
        p += __shfl_xor(p, 8, 32);
        part[r] = p;
    }

    float b2 = a_b2[0];
    float g  = gamma[0];
    float gammaVal = fmaxf(g, 0.0f) + log1pf(expf(-fabsf(g)));   // softplus

    float cc[8];
    float lsum = 0.0f;
    #pragma unroll
    for (int r = 0; r < 8; ++r) {
        float alpha = part[r] + b2;
        float sv = (alpha > LAM) ? (alpha - LAM)
                 : ((alpha < -LAM) ? (alpha + LAM) : 0.0f);
        lsum += fabsf(sv);
        int s = sM[r];   // 0=neg, 1=zero, 2=pos
        cc[r] = (s == 2) ? sv : ((s == 0) ? (-gammaVal * fabsf(sv)) : 0.0f);
    }
    lsum += __shfl_xor(lsum, 16, 32);   // combine both halves; replicated on all lanes
    if (lane == 0) atomicAdd(lossAcc, lsum);

    // cooperative scatter: edge mm's coeff is on lane 0 (mm<8) or lane 16 (mm>=8)
    #pragma unroll
    for (int mm = 0; mm < 16; ++mm) {
        float cm = __shfl(cc[mm & 7], (mm < 8) ? 0 : 16, 32);
        int   rI = __shfl(rowIdx, mm, 32);
        int   cI = __shfl(colIdx, mm, 32);
        if (cm != 0.0f) {
            const float* vs = V   + (size_t)rI * 64;
            float*       dp = agg + (size_t)cI * 64;
            atomicAdd(dp + lane,      cm * vs[lane]);
            atomicAdd(dp + 32 + lane, cm * vs[32 + lane]);
        }
    }
}

// ---------------------------------------------------------------------------
// Final: H_new = agg@Wout + Wout_b + H@Wself + H. One wave per 16-node tile.
// ---------------------------------------------------------------------------
__global__ __launch_bounds__(256)
void final_gemm_kernel(const float* __restrict__ agg,
                       const half_t* __restrict__ Hh,
                       const float* __restrict__ H,
                       const half_t* __restrict__ WoutPack,
                       const half_t* __restrict__ WselfPack,
                       const float* __restrict__ Wout_b,
                       float* __restrict__ out,
                       int n_nodes) {
    int wave = threadIdx.x >> 5;
    int lane = threadIdx.x & 31;
    int n0 = ((int)blockIdx.x * 8 + wave) * 16;
    if (n0 >= n_nodes) return;

    int m = lane & 15;
    const float*  aggRow = agg + (size_t)(n0 + m) * 64;
    const half_t* hRow   = Hh  + (size_t)(n0 + m) * 128;
    int kOff = (lane & 16) ? 8 : 0;

    v16h aA[2];
    #pragma unroll
    for (int kc = 0; kc < 2; ++kc) {
        v16h a;
        #pragma unroll
        for (int i = 0; i < 8; ++i) {
            a[i]     = (half_t)aggRow[kc * 32 + kOff + i];
            a[i + 8] = (half_t)aggRow[kc * 32 + kOff + 16 + i];
        }
        aA[kc] = a;
    }
    v16h aH[4];
    #pragma unroll
    for (int kc = 0; kc < 4; ++kc)
        aH[kc] = load_a_frag(hRow, kc * 32, lane);

    int rOff = (lane & 16) ? 8 : 0;
    #pragma unroll
    for (int ct = 0; ct < 8; ++ct) {
        v8f acc = {};
        acc = wmma_f16(aA[0], load_b_frag(WoutPack, 2, ct, 0, lane), acc);
        acc = wmma_f16(aA[1], load_b_frag(WoutPack, 2, ct, 1, lane), acc);
        #pragma unroll
        for (int kc = 0; kc < 4; ++kc)
            acc = wmma_f16(aH[kc], load_b_frag(WselfPack, 4, ct, kc, lane), acc);
        int col = ct * 16 + m;
        float bias = Wout_b[col];
        #pragma unroll
        for (int r = 0; r < 8; ++r) {
            size_t node = (size_t)(n0 + r + rOff);
            out[node * 128 + col] = acc[r] + bias + H[node * 128 + col];
        }
    }
}

__global__ void loss_final_kernel(const float* __restrict__ lossAcc,
                                  float* __restrict__ out, long pos, float invE) {
    if (blockIdx.x == 0 && threadIdx.x == 0) out[pos] = lossAcc[0] * invE;
}

// ---------------------------------------------------------------------------
extern "C" void kernel_launch(void* const* d_in, const int* in_sizes, int n_in,
                              void* d_out, int out_size, void* d_ws, size_t ws_size,
                              hipStream_t stream) {
    const float* H         = (const float*)d_in[0];
    const int*   edge_idx  = (const int*)  d_in[1];
    const int*   edge_sign = (const int*)  d_in[2];
    const float* W_v       = (const float*)d_in[3];
    const float* W_t       = (const float*)d_in[4];
    const float* sign_emb  = (const float*)d_in[5];
    const float* a_w1      = (const float*)d_in[6];
    const float* a_b1      = (const float*)d_in[7];
    const float* a_w2      = (const float*)d_in[8];
    const float* a_b2      = (const float*)d_in[9];
    const float* W_self    = (const float*)d_in[10];
    const float* W_out_w   = (const float*)d_in[11];
    const float* W_out_b   = (const float*)d_in[12];
    const float* gamma     = (const float*)d_in[13];

    const int n_nodes = in_sizes[0] / 128;
    const int n_edges = in_sizes[2];

    // ---- workspace layout (256B aligned) ----
    char* ws = (char*)d_ws;
    size_t cur = 0;
    auto alloc = [&](size_t bytes) {
        void* p = ws + cur;
        cur += (bytes + 255) & ~(size_t)255;
        return p;
    };
    float*  agg   = (float*) alloc((size_t)n_nodes * 64 * 4);
    float*  lossA = (float*) alloc(4);
    float*  V     = (float*) alloc((size_t)n_nodes * 64 * 4);
    half_t* Vh    = (half_t*)alloc((size_t)n_nodes * 64 * 2);
    half_t* Th    = (half_t*)alloc((size_t)n_nodes * 64 * 2);
    half_t* Hh    = (half_t*)alloc((size_t)n_nodes * 128 * 2);
    half_t* WvP   = (half_t*)alloc(4 * 4 * 512 * 2);   // K=128,N=64
    half_t* WtP   = (half_t*)alloc(4 * 4 * 512 * 2);
    half_t* WsP   = (half_t*)alloc(4 * 8 * 512 * 2);   // K=128,N=128
    half_t* WoP   = (half_t*)alloc(2 * 8 * 512 * 2);   // K=64, N=128
    half_t* A1tP  = (half_t*)alloc(2 * 8 * 512 * 2);   // K=64, N=128
    half_t* A1vP  = (half_t*)alloc(2 * 8 * 512 * 2);
    float*  sembC = (float*) alloc(3 * 128 * 4);

    // zero agg + loss accumulator (contiguous region)
    hipMemsetAsync(agg, 0, ((size_t)n_nodes * 64 * 4 + 255 + 256) & ~(size_t)255, stream);

    // precompute / repack
    long hElems = (long)n_nodes * 128;
    cvt_h_kernel<<<2048, 256, 0, stream>>>(H, Hh, hElems);
    semb_kernel<<<2, 256, 0, stream>>>(sign_emb, a_w1, a_b1, sembC);
    pack_b_kernel<<<32, 256, 0, stream>>>(W_v,     64,  0,  128, 64,  WvP);
    pack_b_kernel<<<32, 256, 0, stream>>>(W_t,     64,  0,  128, 64,  WtP);
    pack_b_kernel<<<64, 256, 0, stream>>>(W_self,  128, 0,  128, 128, WsP);
    pack_b_kernel<<<32, 256, 0, stream>>>(W_out_w, 128, 0,  64,  128, WoP);
    pack_b_kernel<<<32, 256, 0, stream>>>(a_w1,    128, 0,  64,  128, A1tP);
    pack_b_kernel<<<32, 256, 0, stream>>>(a_w1,    128, 64, 64,  128, A1vP);

    // node projections
    int nodeTiles = (n_nodes + 15) / 16;
    node_gemm_kernel<<<nodeTiles, 256, 0, stream>>>(Hh, WvP, WtP, V, Vh, Th, n_nodes);

    // edge MLP + scatter
    int edgeTiles = (n_edges + 15) / 16;
    int edgeBlocks = (edgeTiles + 7) / 8;
    edge_mlp_kernel<<<edgeBlocks, 256, 0, stream>>>(
        edge_idx, edge_sign, Th, Vh, V, A1tP, A1vP, sembC,
        a_w2, a_b2, gamma, agg, lossA, n_edges);

    // output GEMM + residual
    int finBlocks = (nodeTiles + 7) / 8;
    final_gemm_kernel<<<finBlocks, 256, 0, stream>>>(
        agg, Hh, H, WoP, WsP, W_out_b, (float*)d_out, n_nodes);

    loss_final_kernel<<<1, 32, 0, stream>>>(
        lossA, (float*)d_out, (long)n_nodes * 128, 1.0f / (float)n_edges);
}
